// TFN_Tetris_Model_56410100466292
// MI455X (gfx1250) — matrix operand order repaired and checked by reading, not compiled, not run on hardware
//
#include <hip/hip_runtime.h>
#include <math.h>

// ---------------------------------------------------------------------------
// TFN (Tensor Field Network) fused implementation for gfx1250 (MI455X).
// Radial MLPs run on V_WMMA_F32_16X16X32_BF16 (bf16 in, f32 accumulate).
// ---------------------------------------------------------------------------

typedef __attribute__((ext_vector_type(16))) __bf16 v16bf;
typedef __attribute__((ext_vector_type(8)))  float  v8f;

#define NPTS 1024
#define CCH  16
#define RCNT 32
#define EPSV 1e-8f
#define WVB  8              // waves per block (256 threads, wave32)
#define NTILE (NPTS / 16)   // 64 b-tiles of 16 pairs

#define RBF_STEP  (3.5f / 31.0f)     // linspace(0, 3.5, 32) spacing
#define RBF_GAMMA (32.0f / 3.5f)

__device__ __forceinline__ float eluf(float x) {
    return x > 0.f ? x : (__expf(x) - 1.f);
}

// sum across the 16 pair-lanes inside each half-wave (xor masks keep bit4)
__device__ __forceinline__ float wred16(float v) {
    v += __shfl_xor(v, 1, 32);
    v += __shfl_xor(v, 2, 32);
    v += __shfl_xor(v, 4, 32);
    v += __shfl_xor(v, 8, 32);
    return v;
}
__device__ __forceinline__ float wred32(float v) {
    v = wred16(v);
    v += __shfl_xor(v, 16, 32);
    return v;
}

// ---------------------------------------------------------------------------
// Layer 1: only degree-0 input (== ones). Per point a:
//   out0[a]    = sum_b r00(a,b)                      (scalar radial, F=1)
//   out1[a,i]  = sum_b r01(a,b) * unit(a,b,i)        (diag killed by unit==0)
// then self-interaction + nonlinearity -> x0_l1[N,16], x1_l1[N,16,3].
// One wave per 'a'; WMMA over 16-pair b-tiles:
//   hidden^T[32h x 16p] = W1^T(A, two 16-row halves) @ rbf^T(B)
// Second GEMM (F=1) degenerates to a per-lane dot with W2.
// ---------------------------------------------------------------------------
__global__ __launch_bounds__(256) void tfn_layer1(
    const float* __restrict__ pts,
    const float* __restrict__ W1, const float* __restrict__ B1,
    const float* __restrict__ W2, const float* __restrict__ B2,
    const float* __restrict__ si0w, const float* __restrict__ si0b,
    const float* __restrict__ si1w, const float* __restrict__ nlb,
    float* __restrict__ x0o, float* __restrict__ x1o)
{
    __shared__ float4 spts[NPTS];
    for (int i = threadIdx.x; i < NPTS; i += blockDim.x)
        spts[i] = make_float4(pts[3 * i], pts[3 * i + 1], pts[3 * i + 2], 0.f);
    __syncthreads();

    const int lane = threadIdx.x & 31;
    const int wid  = threadIdx.x >> 5;
    const int a    = blockIdx.x * WVB + wid;
    const int lh   = lane >> 4;     // lane half
    const int p    = lane & 15;     // pair slot within tile

    // Preload A fragments (W1^T rows m / 16+m) and per-lane bias/W2 slices.
    // Every loop fully unrolled so fragments stay in VGPRs (no scratch).
    v16bf aF[2][2];
    float b1l[2][8], b1h[2][8], w2l[2][8], w2h[2][8], b2s[2];
#pragma unroll
    for (int f = 0; f < 2; ++f) {
        const float* w1 = W1 + f * RCNT * RCNT;   // [kin][kout]
#pragma unroll
        for (int e = 0; e < 16; ++e) {
            int K = ((e >> 3) << 4) + (lh << 3) + (e & 7);  // A-layout K index
            aF[f][0][e] = (__bf16)w1[K * RCNT + p];
            aF[f][1][e] = (__bf16)w1[K * RCNT + 16 + p];
        }
#pragma unroll
        for (int v = 0; v < 8; ++v) {
            b1l[f][v] = B1[f * RCNT + v + 8 * lh];
            b1h[f][v] = B1[f * RCNT + 16 + v + 8 * lh];
            w2l[f][v] = W2[f * RCNT + v + 8 * lh];
            w2h[f][v] = W2[f * RCNT + 16 + v + 8 * lh];
        }
        b2s[f] = B2[f];
    }

    const float4 pa = spts[a];
    float acc0 = 0.f, a1x = 0.f, a1y = 0.f, a1z = 0.f;

#pragma unroll 1
    for (int tb = 0; tb < NTILE; ++tb) {
        const int b = tb * 16 + p;
        const float4 pb = spts[b];
        float dx = pa.x - pb.x, dy = pa.y - pb.y, dz = pa.z - pb.z;
        float ss  = dx * dx + dy * dy + dz * dz;
        float dij = (ss < EPSV) ? 0.f : sqrtf(ss);
        float inl = rsqrtf(fmaxf(ss, EPSV));       // 1/norm_eps
        float ux = dx * inl, uy = dy * inl, uz = dz * inl;

        // rbf B fragment: lane covers K = 16*(e>>3) + 8*lh + (e&7) for its pair column p
        v16bf bF;
#pragma unroll
        for (int e = 0; e < 16; ++e) {
            float c = (float)(((e >> 3) << 4) + (lh << 3) + (e & 7)) * RBF_STEP;
            float t = dij - c;
            bF[e] = (__bf16)__expf(-RBF_GAMMA * t * t);
        }

#pragma unroll
        for (int f = 0; f < 2; ++f) {
            v8f cz = {};
            v8f d0 = __builtin_amdgcn_wmma_f32_16x16x32_bf16(
                false, aF[f][0], false, bF, (short)0, cz, false, false);
            v8f d1 = __builtin_amdgcn_wmma_f32_16x16x32_bf16(
                false, aF[f][1], false, bF, (short)0, cz, false, false);
            float partial = 0.f;
#pragma unroll
            for (int v = 0; v < 8; ++v) {
                float h0 = fmaxf(d0[v] + b1l[f][v], 0.f);
                float h1 = fmaxf(d1[v] + b1h[f][v], 0.f);
                partial += h0 * w2l[f][v] + h1 * w2h[f][v];
            }
            // lanes L, L+16 hold complementary h halves of the same pair
            float outp = partial + __shfl_xor(partial, 16, 32) + b2s[f];
            if (f == 0) {
                acc0 += outp;
            } else {
                a1x += outp * ux; a1y += outp * uy; a1z += outp * uz;
            }
        }
    }

    // both halves accumulated identical pair values -> *0.5 after full reduce
    float o0  = wred32(acc0) * 0.5f;
    float o1x = wred32(a1x) * 0.5f;
    float o1y = wred32(a1y) * 0.5f;
    float o1z = wred32(a1z) * 0.5f;

    if (lane < CCH) {
        const int g = lane;
        float xg = eluf(o0 * si0w[g] + si0b[g]);
        float w  = si1w[g];
        float vx = o1x * w, vy = o1y * w, vz = o1z * w;
        float n  = sqrtf(fmaxf(vx * vx + vy * vy + vz * vz, EPSV));
        float sc = eluf(n + nlb[g]) / n;
        x0o[a * CCH + g] = xg;
        x1o[(a * CCH + g) * 3 + 0] = vx * sc;
        x1o[(a * CCH + g) * 3 + 1] = vy * sc;
        x1o[(a * CCH + g) * 3 + 2] = vz * sc;
    }
}

// ---------------------------------------------------------------------------
// Layer 2: only the degree-0 outputs survive to the readout, so we need
// just filters 0 (0x0->0) and 2 (1x1->0):
//   c00[a,f] = sum_b rad0(a,b,f) * x0[b,f]
//   c10[a,f] = sum_b rad2(a,b,f) * dot(unit(a,b), x1[b,f,:])
// Per 16-pair tile / filter: 2 WMMAs (hidden) -> relu/bias -> half-wave
// swap + bf16 pack -> 1 WMMA (out^T = W2^T @ hidden^T) -> per-lane epilogue.
// Per-lane x0/x1 slices are contiguous (f = 8*lh + v) -> float4 loads,
// issued before the WMMA chain so global latency hides under the GEMMs.
// ---------------------------------------------------------------------------
__global__ __launch_bounds__(256) void tfn_layer2(
    const float* __restrict__ pts,
    const float* __restrict__ W1, const float* __restrict__ B1,
    const float* __restrict__ W2, const float* __restrict__ B2,
    const float* __restrict__ si0w, const float* __restrict__ si0b,
    const float* __restrict__ x0i, const float* __restrict__ x1i,
    float* __restrict__ x0l2)
{
    __shared__ float4 spts[NPTS];
    __shared__ float cat_sh[WVB][32];
    for (int i = threadIdx.x; i < NPTS; i += blockDim.x)
        spts[i] = make_float4(pts[3 * i], pts[3 * i + 1], pts[3 * i + 2], 0.f);
    __syncthreads();

    const int lane = threadIdx.x & 31;
    const int wid  = threadIdx.x >> 5;
    const int a    = blockIdx.x * WVB + wid;
    const int lh   = lane >> 4;
    const int p    = lane & 15;

    const int fidx[2] = {0, 2};   // filters used (0x0->0, 1x1->0)
    v16bf aW1[2][2], aW2[2];
    float b1l[2][8], b1h[2][8], b2v[2][8];
#pragma unroll
    for (int q = 0; q < 2; ++q) {
        const int f = fidx[q];
        const float* w1 = W1 + f * RCNT * RCNT;   // [kin][kout]
        const float* w2 = W2 + f * RCNT * CCH;    // [kin][fout]
#pragma unroll
        for (int e = 0; e < 16; ++e) {
            int K = ((e >> 3) << 4) + (lh << 3) + (e & 7);
            aW1[q][0][e] = (__bf16)w1[K * RCNT + p];
            aW1[q][1][e] = (__bf16)w1[K * RCNT + 16 + p];
            aW2[q][e]    = (__bf16)w2[(16 * lh + e) * CCH + p];     // W2^T[m=p][K]
        }
#pragma unroll
        for (int v = 0; v < 8; ++v) {
            b1l[q][v] = B1[f * RCNT + v + 8 * lh];
            b1h[q][v] = B1[f * RCNT + 16 + v + 8 * lh];
            b2v[q][v] = B2[f * CCH + v + 8 * lh];
        }
    }

    const float4 pa = spts[a];

    float acc00[8], acc10[8];
#pragma unroll
    for (int v = 0; v < 8; ++v) { acc00[v] = 0.f; acc10[v] = 0.f; }

#pragma unroll 1
    for (int tb = 0; tb < NTILE; ++tb) {
        const int b = tb * 16 + p;

        // ---- issue global loads for this tile's x0/x1 slices first ----
        // lane covers f = 8*lh + 0..7 -> 8 contiguous floats of x0,
        // 24 contiguous floats of x1 (both 16B aligned).
        const float4* x0v = (const float4*)(x0i + b * CCH + 8 * lh);
        const float4* x1v = (const float4*)(x1i + (b * CCH + 8 * lh) * 3);
        float x0r[8], x1r[24];
        {
            float4 t0 = x0v[0], t1 = x0v[1];
            x0r[0] = t0.x; x0r[1] = t0.y; x0r[2] = t0.z; x0r[3] = t0.w;
            x0r[4] = t1.x; x0r[5] = t1.y; x0r[6] = t1.z; x0r[7] = t1.w;
        }
#pragma unroll
        for (int w = 0; w < 6; ++w) {
            float4 t = x1v[w];
            x1r[4 * w + 0] = t.x; x1r[4 * w + 1] = t.y;
            x1r[4 * w + 2] = t.z; x1r[4 * w + 3] = t.w;
        }

        // ---- geometry + rbf fragment ----
        const float4 pb = spts[b];
        float dx = pa.x - pb.x, dy = pa.y - pb.y, dz = pa.z - pb.z;
        float ss  = dx * dx + dy * dy + dz * dz;
        float dij = (ss < EPSV) ? 0.f : sqrtf(ss);
        float inl = rsqrtf(fmaxf(ss, EPSV));
        float ux = dx * inl, uy = dy * inl, uz = dz * inl;

        v16bf bF;
#pragma unroll
        for (int e = 0; e < 16; ++e) {
            float c = (float)(((e >> 3) << 4) + (lh << 3) + (e & 7)) * RBF_STEP;
            float t = dij - c;
            bF[e] = (__bf16)__expf(-RBF_GAMMA * t * t);
        }

#pragma unroll
        for (int q = 0; q < 2; ++q) {
            v8f cz = {};
            v8f d0 = __builtin_amdgcn_wmma_f32_16x16x32_bf16(
                false, aW1[q][0], false, bF, (short)0, cz, false, false);
            v8f d1 = __builtin_amdgcn_wmma_f32_16x16x32_bf16(
                false, aW1[q][1], false, bF, (short)0, cz, false, false);

            // bias + relu, then half-wave swap to build hidden^T B fragment
            float h0[8], h1[8], lo[8], hi[8];
#pragma unroll
            for (int v = 0; v < 8; ++v) {
                h0[v] = fmaxf(d0[v] + b1l[q][v], 0.f);
                h1[v] = fmaxf(d1[v] + b1h[q][v], 0.f);
            }
#pragma unroll
            for (int v = 0; v < 8; ++v) {
                float x0s = __shfl_xor(h0[v], 16, 32);
                float x1s = __shfl_xor(h1[v], 16, 32);
                lo[v] = lh ? x1s : h0[v];   // rows K = 16*lh + 0..7
                hi[v] = lh ? h1[v] : x0s;   // rows K = 16*lh + 8..15
            }
            v16bf bH;
#pragma unroll
            for (int e = 0; e < 8; ++e) {
                bH[e]     = (__bf16)lo[e];
                bH[e + 8] = (__bf16)hi[e];
            }
            v8f d2 = __builtin_amdgcn_wmma_f32_16x16x32_bf16(
                false, aW2[q], false, bH, (short)0, cz, false, false);

            // out^T: lane col = pair p (-> b), row f = v + 8*lh
            if (q == 0) {
#pragma unroll
                for (int v = 0; v < 8; ++v)
                    acc00[v] += (d2[v] + b2v[0][v]) * x0r[v];
            } else {
#pragma unroll
                for (int v = 0; v < 8; ++v) {
                    float dt = ux * x1r[3 * v] + uy * x1r[3 * v + 1]
                             + uz * x1r[3 * v + 2];
                    acc10[v] += (d2[v] + b2v[1][v]) * dt;
                }
            }
        }
    }

#pragma unroll
    for (int v = 0; v < 8; ++v) {
        acc00[v] = wred16(acc00[v]);
        acc10[v] = wred16(acc10[v]);
    }
    if (p == 0) {   // lanes 0 and 16 publish their f-halves
#pragma unroll
        for (int v = 0; v < 8; ++v) {
            cat_sh[wid][lh * 8 + v]      = acc00[v];   // cat0[0..15]  = c00
            cat_sh[wid][16 + lh * 8 + v] = acc10[v];   // cat0[16..31] = c10
        }
    }
    __syncthreads();

    if (lane < CCH) {
        const int g = lane;
        float s = si0b[g];
#pragma unroll
        for (int f = 0; f < 32; ++f) s += si0w[g * 32 + f] * cat_sh[wid][f];
        x0l2[a * CCH + g] = eluf(s);
    }
}

// ---------------------------------------------------------------------------
// Readout: deterministic mean-pool over points, then 16->8 FC.
// 256 threads: thread (chunk,g) sums 64 points, LDS tree-reduce.
// ---------------------------------------------------------------------------
__global__ __launch_bounds__(256) void tfn_readout(
    const float* __restrict__ x0l2,
    const float* __restrict__ fcw,
    const float* __restrict__ fcb,
    float* __restrict__ out)
{
    __shared__ float part[256];
    const int t = threadIdx.x;
    const int g = t & 15;
    const int chunk = t >> 4;        // 16 chunks of 64 points

    float s = 0.f;
#pragma unroll 1
    for (int k = 0; k < NPTS / 16; ++k)
        s += x0l2[(chunk * (NPTS / 16) + k) * CCH + g];
    part[t] = s;
    __syncthreads();

    if (t < CCH) {
        float m = 0.f;
#pragma unroll
        for (int c = 0; c < 16; ++c) m += part[c * 16 + t];
        part[t] = m * (1.0f / NPTS);
    }
    __syncthreads();

    if (t < 8) {
        float s2 = fcb[t];
#pragma unroll
        for (int gg = 0; gg < CCH; ++gg) s2 += part[gg] * fcw[gg * 8 + t];
        out[t] = s2;
    }
}

extern "C" void kernel_launch(void* const* d_in, const int* in_sizes, int n_in,
                              void* d_out, int out_size, void* d_ws, size_t ws_size,
                              hipStream_t stream)
{
    const float* pts     = (const float*)d_in[0];
    const float* l1_W1   = (const float*)d_in[1];
    const float* l1_B1   = (const float*)d_in[2];
    const float* l1_W2   = (const float*)d_in[3];
    const float* l1_B2   = (const float*)d_in[4];
    const float* l1_si0w = (const float*)d_in[5];
    const float* l1_si0b = (const float*)d_in[6];
    const float* l1_si1w = (const float*)d_in[7];
    const float* l1_nlb  = (const float*)d_in[8];
    const float* l2_W1   = (const float*)d_in[9];
    const float* l2_B1   = (const float*)d_in[10];
    const float* l2_W2   = (const float*)d_in[11];
    const float* l2_B2   = (const float*)d_in[12];
    const float* l2_si0w = (const float*)d_in[13];
    const float* l2_si0b = (const float*)d_in[14];
    // d_in[15] l2_si1_w, d_in[16] l2_nl_b: dead (layer-2 x1 unused by readout)
    const float* fc_w    = (const float*)d_in[17];
    const float* fc_b    = (const float*)d_in[18];

    float* x0b  = (float*)d_ws;                  // [1024*16]
    float* x1b  = x0b + NPTS * CCH;              // [1024*16*3]
    float* x0l2 = x1b + NPTS * CCH * 3;          // [1024*16]

    tfn_layer1<<<NPTS / WVB, 256, 0, stream>>>(
        pts, l1_W1, l1_B1, l1_W2, l1_B2,
        l1_si0w, l1_si0b, l1_si1w, l1_nlb, x0b, x1b);

    tfn_layer2<<<NPTS / WVB, 256, 0, stream>>>(
        pts, l2_W1, l2_B1, l2_W2, l2_B2,
        l2_si0w, l2_si0b, x0b, x1b, x0l2);

    tfn_readout<<<1, 256, 0, stream>>>(x0l2, fc_w, fc_b, (float*)d_out);
}
